// AttentionCompetition_72155450573314
// MI455X (gfx1250) — matrix operand order repaired and checked by reading, not compiled, not run on hardware
//
#include <hip/hip_runtime.h>
#include <hip/hip_bf16.h>

typedef __attribute__((ext_vector_type(16))) _Float16     v16h;
typedef __attribute__((ext_vector_type(8)))  _Float16     v8h;
typedef __attribute__((ext_vector_type(8)))  float        v8f;
typedef __attribute__((ext_vector_type(4)))  unsigned int u32x4;
typedef __attribute__((ext_vector_type(8)))  unsigned int u32x8;

__device__ __forceinline__ v16h cat16(v8h lo, v8h hi) {
  return __builtin_shufflevector(lo, hi, 0,1,2,3,4,5,6,7,8,9,10,11,12,13,14,15);
}
__device__ __forceinline__ v16h ld_frag(const _Float16* p0, const _Float16* p1) {
  return cat16(*(const v8h*)p0, *(const v8h*)p1);
}
__device__ __forceinline__ v8f wmma_f16(v16h a, v16h b, v8f c) {
  return __builtin_amdgcn_wmma_f32_16x16x32_f16(false, a, false, b, (short)0, c, false, false);
}

// -------- Tensor Data Mover: async 2D tile (f16) global -> LDS, padded rows --
// D# group0: count=1 | lds_addr | global_addr(57b) | type=2
// D# group1: data_size=2B, pad_enable, pad_interval=16 DW, pad_amount=4 DW
//            (64B data row + 16B pad = 80B stride = LDSTR(40) halves),
//            tensor dims = tile dims, row stride = K elements.
__device__ __forceinline__ unsigned lds_off(const void* p) {
  return (unsigned)(size_t)p;   // LDS aperture: addr[31:0] == LDS byte offset
}
__device__ __forceinline__ void tdm_load_tile_f16(const _Float16* gaddr,
                                                  unsigned ldsAddr,
                                                  unsigned strideElems,
                                                  unsigned tileD0,
                                                  unsigned tileD1) {
  unsigned long long ga = (unsigned long long)(size_t)gaddr;
  u32x4 g0;
  g0[0] = 1u;                                   // count=1 (valid descriptor)
  g0[1] = ldsAddr;                              // lds_addr
  g0[2] = (unsigned)ga;                         // global_addr[31:0]
  g0[3] = (unsigned)(ga >> 32) | (2u << 30);    // global_addr[56:32] | type=2
  u32x8 g1;
  g1[0] = (1u << 16)        // data_size = 2 bytes
        | (1u << 20)        // pad_enable
        | (3u << 22)        // pad_interval: 16 DWORDs (64B)
        | (3u << 25);       // pad_amount:   4 DWORDs (16B)
  g1[1] = (tileD0 & 0xFFFFu) << 16;                       // tensor_dim0 lo
  g1[2] = (tileD0 >> 16) | ((tileD1 & 0xFFFFu) << 16);    // dim0 hi | dim1 lo
  g1[3] = (tileD1 >> 16) | (tileD0 << 16);                // dim1 hi | tile_dim0
  g1[4] = tileD1 & 0xFFFFu;                               // tile_dim1 (dim2=0)
  g1[5] = strideElems;                                    // tensor_dim0_stride
  g1[6] = 0u;
  g1[7] = 0u;
  asm volatile("tensor_load_to_lds %0, %1" :: "s"(g0), "s"(g1) : "memory");
}
__device__ __forceinline__ void wait_tensorcnt0() {
#if __has_builtin(__builtin_amdgcn_s_wait_tensorcnt)
  __builtin_amdgcn_s_wait_tensorcnt(0);
#else
  asm volatile("s_wait_tensorcnt 0x0" ::: "memory");
#endif
}
__device__ __forceinline__ void wait_tensorcnt2() {
#if __has_builtin(__builtin_amdgcn_s_wait_tensorcnt)
  __builtin_amdgcn_s_wait_tensorcnt(2);
#else
  asm volatile("s_wait_tensorcnt 0x2" ::: "memory");
#endif
}

// ---------------------------------------------------------------- converts
__global__ void f32_to_f16_kernel(const float* __restrict__ src,
                                  _Float16* __restrict__ dst, int n) {
  int i = blockIdx.x * blockDim.x + threadIdx.x;
  if (i < n) dst[i] = (_Float16)src[i];
}

// ---------------------------------------------------------------- WMMA GEMM
// D[m,n] = sum_k A[m,k] * W[n,k] + bias[n]    (A: MxK f16, W: NxK f16)
// TDM double-buffered LDS staging; wave 0 drives the DMA.
// mode 0: QKV scatter into per-head Q (p0), K (p1), V-transposed (p2), f16
// mode 1: write fp32 (p0) and f16 (p1) row-major MxN
// mode 2: relu then f16 (p0) row-major MxN
#define BM 128
#define BN 64
#define BK 32
#define LDSTR 40   // 32 halves + 8-halves pad (80B stride; matches TDM pad)

__global__ __launch_bounds__(256, 2)
void gemm_wmma(const _Float16* __restrict__ A, const _Float16* __restrict__ W,
               const float* __restrict__ bias, int M, int N, int K, int mode,
               void* p0, void* p1, void* p2) {
  __shared__ __align__(16) _Float16 As[2][BM * LDSTR];
  __shared__ __align__(16) _Float16 Bs[2][BN * LDSTR];

  const int tid  = threadIdx.x;
  const int lane = tid & 31;
  const int wv   = tid >> 5;        // 0..7
  const int mw   = wv & 3;          // 4 wave-rows
  const int nw   = wv >> 2;         // 2 wave-cols
  const int l16  = lane & 15;
  const int lh   = lane >> 4;
  const int m0   = blockIdx.y * BM;
  const int n0   = blockIdx.x * BN;
  const bool dmaWave = (wv == 0);

  v8f acc[2][2];
  #pragma unroll
  for (int j = 0; j < 2; ++j) {
    int col = n0 + nw * 32 + j * 16 + l16;
    float bv = bias[col];
    #pragma unroll
    for (int i = 0; i < 2; ++i) {
      #pragma unroll
      for (int r = 0; r < 8; ++r) acc[i][j][r] = bv;
    }
  }

  // prologue: DMA first K-slab into buffer 0
  if (dmaWave) {
    tdm_load_tile_f16(A + (size_t)m0 * K, lds_off(&As[0][0]), K, BK, BM);
    tdm_load_tile_f16(W + (size_t)n0 * K, lds_off(&Bs[0][0]), K, BK, BN);
  }

  int buf = 0;
  for (int k0 = 0; k0 < K; k0 += BK) {
    const bool hasNext = (k0 + BK) < K;
    if (dmaWave) {
      if (hasNext) {  // kick next slab into the other buffer, then wait current
        tdm_load_tile_f16(A + (size_t)m0 * K + (k0 + BK),
                          lds_off(&As[buf ^ 1][0]), K, BK, BM);
        tdm_load_tile_f16(W + (size_t)n0 * K + (k0 + BK),
                          lds_off(&Bs[buf ^ 1][0]), K, BK, BN);
        wait_tensorcnt2();          // the 2 current-buffer DMAs are done
      } else {
        wait_tensorcnt0();
      }
    }
    __syncthreads();                // current buffer ready for all waves

    v16h aF[2], bF[2];
    #pragma unroll
    for (int i = 0; i < 2; ++i) {
      int row = mw * 32 + i * 16 + l16;
      int kb = lh * 8;
      aF[i] = ld_frag(&As[buf][row * LDSTR + kb], &As[buf][row * LDSTR + kb + 16]);
    }
    #pragma unroll
    for (int j = 0; j < 2; ++j) {
      int col = nw * 32 + j * 16 + l16;
      int kb = lh * 8;
      bF[j] = ld_frag(&Bs[buf][col * LDSTR + kb], &Bs[buf][col * LDSTR + kb + 16]);
    }
    #pragma unroll
    for (int i = 0; i < 2; ++i)
      #pragma unroll
      for (int j = 0; j < 2; ++j)
        acc[i][j] = wmma_f16(aF[i], bF[j], acc[i][j]);

    __syncthreads();                // reads done before this buffer is re-filled
    buf ^= 1;
  }

  // epilogue
  #pragma unroll
  for (int i = 0; i < 2; ++i) {
    #pragma unroll
    for (int j = 0; j < 2; ++j) {
      int col   = n0 + nw * 32 + j * 16 + l16;
      int rbase = m0 + mw * 32 + i * 16 + lh * 8;
      #pragma unroll
      for (int r = 0; r < 8; ++r) {
        int row = rbase + r;
        float v = acc[i][j][r];
        if (mode == 0) {
          int sect = col >> 9, jj = col & 511;
          int h = jj >> 6, d = jj & 63;
          int b = row >> 8, n = row & 255;
          size_t bh = (size_t)(b * 8 + h);
          _Float16 hv = (_Float16)v;
          if (sect == 0)      ((_Float16*)p0)[(bh * 256 + n) * 64 + d] = hv;
          else if (sect == 1) ((_Float16*)p1)[(bh * 256 + n) * 64 + d] = hv;
          else                ((_Float16*)p2)[(bh * 64 + d) * 256 + n] = hv;
        } else if (mode == 1) {
          ((float*)p0)[(size_t)row * N + col] = v;
          ((_Float16*)p1)[(size_t)row * N + col] = (_Float16)v;
        } else {
          float rv = v > 0.f ? v : 0.f;
          ((_Float16*)p0)[(size_t)row * N + col] = (_Float16)rv;
        }
      }
    }
  }
}

// ---------------------------------------------------------------- flash attention
// 1 wave per (b,h, 16-row q block). Q/K: (bh,256,64) f16, Vt: (bh,64,256) f16.
__global__ __launch_bounds__(32)
void attn_flash(const _Float16* __restrict__ Qh, const _Float16* __restrict__ Kh,
                const _Float16* __restrict__ Vt, _Float16* __restrict__ Ctxh) {
  __shared__ __align__(16) _Float16 Ps[16 * LDSTR];
  const int lane = threadIdx.x & 31;
  const int l16 = lane & 15, lh = lane >> 4;
  const int bh = blockIdx.x >> 4;
  const int qt = blockIdx.x & 15;

  const _Float16* Qb = Qh + (size_t)bh * 256 * 64;
  const _Float16* Kb = Kh + (size_t)bh * 256 * 64;
  const _Float16* Vb = Vt + (size_t)bh * 64 * 256;

  int qrow = qt * 16 + l16;
  v16h qA[2];
  #pragma unroll
  for (int t = 0; t < 2; ++t) {
    const _Float16* p = Qb + (size_t)qrow * 64 + t * 32 + lh * 8;
    qA[t] = ld_frag(p, p + 16);
  }

  v8f o[4];
  float mrow[8], lrow[8];
  #pragma unroll
  for (int r = 0; r < 8; ++r) {
    mrow[r] = -3.0e38f; lrow[r] = 0.f;
    #pragma unroll
    for (int j = 0; j < 4; ++j) o[j][r] = 0.f;
  }

  for (int it = 0; it < 8; ++it) {
    int kBase = it * 32;
    if (it < 7) {  // prefetch next slab of K rows and V columns
      __builtin_prefetch(Kb + (size_t)(kBase + 32 + l16) * 64, 0, 1);
      __builtin_prefetch(Vb + (size_t)(l16 * 2 + lh) * 256 + kBase + 32, 0, 1);
    }
    v8f s[2];
    #pragma unroll
    for (int tile = 0; tile < 2; ++tile) {
      int krow = kBase + tile * 16 + l16;
      v8f a;
      #pragma unroll
      for (int r = 0; r < 8; ++r) a[r] = 0.f;
      #pragma unroll
      for (int t = 0; t < 2; ++t) {
        const _Float16* p = Kb + (size_t)krow * 64 + t * 32 + lh * 8;
        a = wmma_f16(qA[t], ld_frag(p, p + 16), a);
      }
      s[tile] = a;
    }
    #pragma unroll
    for (int r = 0; r < 8; ++r) {
      float s0 = s[0][r] * 0.125f, s1 = s[1][r] * 0.125f;
      float mx = fmaxf(s0, s1);
      #pragma unroll
      for (int off = 8; off >= 1; off >>= 1) mx = fmaxf(mx, __shfl_xor(mx, off, 32));
      float mn = fmaxf(mrow[r], mx);
      float sc = __expf(mrow[r] - mn);
      float p0 = __expf(s0 - mn), p1 = __expf(s1 - mn);
      float ps = p0 + p1;
      #pragma unroll
      for (int off = 8; off >= 1; off >>= 1) ps += __shfl_xor(ps, off, 32);
      lrow[r] = lrow[r] * sc + ps;
      mrow[r] = mn;
      #pragma unroll
      for (int j = 0; j < 4; ++j) o[j][r] *= sc;
      s[0][r] = p0; s[1][r] = p1;
    }
    __syncthreads();
    #pragma unroll
    for (int r = 0; r < 8; ++r) {
      int prow = r + lh * 8;
      Ps[prow * LDSTR + l16]      = (_Float16)s[0][r];
      Ps[prow * LDSTR + 16 + l16] = (_Float16)s[1][r];
    }
    __syncthreads();
    v16h pA = ld_frag(&Ps[l16 * LDSTR + lh * 8], &Ps[l16 * LDSTR + lh * 8 + 16]);
    #pragma unroll
    for (int j = 0; j < 4; ++j) {
      const _Float16* p = Vb + (size_t)(j * 16 + l16) * 256 + kBase + lh * 8;
      o[j] = wmma_f16(pA, ld_frag(p, p + 16), o[j]);
    }
    __syncthreads();
  }
  int b = bh >> 3, h = bh & 7;
  #pragma unroll
  for (int r = 0; r < 8; ++r) {
    float inv = 1.0f / lrow[r];
    int m = qt * 16 + r + lh * 8;
    size_t base = ((size_t)(b * 256 + m)) * 512 + h * 64;
    #pragma unroll
    for (int j = 0; j < 4; ++j)
      Ctxh[base + j * 16 + l16] = (_Float16)(o[j][r] * inv);
  }
}

// ---------------------------------------------------------------- gate dot + softmax + top-7
__global__ __launch_bounds__(256)
void gate_topk(const _Float16* __restrict__ Hmat, const float* __restrict__ g2w,
               const float* __restrict__ g2b, const float* __restrict__ sal,
               float* __restrict__ wOut) {
  const int b = blockIdx.x, n = threadIdx.x;
  __shared__ float red[256];
  __shared__ int   redi[256];

  const _Float16* hrow = Hmat + ((size_t)(b * 256 + n)) * 256;
  float dot = 0.f;
  for (int c = 0; c < 256; ++c) dot += (float)hrow[c] * g2w[c];
  float comb = dot + g2b[0] + sal[b * 256 + n];

  red[n] = comb; __syncthreads();
  for (int s = 128; s > 0; s >>= 1) { if (n < s) red[n] = fmaxf(red[n], red[n + s]); __syncthreads(); }
  float mx = red[0]; __syncthreads();
  float e = __expf(comb - mx);
  red[n] = e; __syncthreads();
  for (int s = 128; s > 0; s >>= 1) { if (n < s) red[n] += red[n + s]; __syncthreads(); }
  float wv = e / red[0]; __syncthreads();

  float cur = wv;
  int selected = 0;
  for (int t = 0; t < 7; ++t) {
    red[n] = cur; redi[n] = n; __syncthreads();
    for (int s = 128; s > 0; s >>= 1) {
      if (n < s) {
        if (red[n + s] > red[n] || (red[n + s] == red[n] && redi[n + s] < redi[n])) {
          red[n] = red[n + s]; redi[n] = redi[n + s];
        }
      }
      __syncthreads();
    }
    int win = redi[0]; __syncthreads();
    if (n == win) { selected = 1; cur = -1.0f; }
    __syncthreads();
  }
  red[n] = selected ? wv : 0.f; __syncthreads();
  for (int s = 128; s > 0; s >>= 1) { if (n < s) red[n] += red[n + s]; __syncthreads(); }
  wOut[b * 256 + n] = (selected ? wv : 0.f) / (red[0] + 1e-8f);
}

// ---------------------------------------------------------------- weight + LayerNorm
__global__ __launch_bounds__(128)
void weight_ln(const float* __restrict__ att, const float* __restrict__ wsel,
               const float* __restrict__ lnw, const float* __restrict__ lnb,
               float* __restrict__ out) {
  const int row = blockIdx.x, t = threadIdx.x;
  __shared__ float rs[128], rs2[128];
  float wv = wsel[row];
  const float* arow = att + (size_t)row * 512;
  float x[4], s = 0.f, s2 = 0.f;
  #pragma unroll
  for (int i = 0; i < 4; ++i) {
    float v = arow[t + i * 128] * wv;
    x[i] = v; s += v; s2 += v * v;
  }
  rs[t] = s; rs2[t] = s2; __syncthreads();
  for (int st = 64; st > 0; st >>= 1) {
    if (t < st) { rs[t] += rs[t + st]; rs2[t] += rs2[t + st]; }
    __syncthreads();
  }
  float mu = rs[0] * (1.0f / 512.f);
  float var = rs2[0] * (1.0f / 512.f) - mu * mu;
  float inv = rsqrtf(var + 1e-5f);
  float* orow = out + (size_t)row * 512;
  #pragma unroll
  for (int i = 0; i < 4; ++i) {
    int c = t + i * 128;
    orow[c] = (x[i] - mu) * inv * lnw[c] + lnb[c];
  }
}

// ---------------------------------------------------------------- launch
extern "C" void kernel_launch(void* const* d_in, const int* in_sizes, int n_in,
                              void* d_out, int out_size, void* d_ws, size_t ws_size,
                              hipStream_t stream) {
  const float* features = (const float*)d_in[0];
  const float* sal      = (const float*)d_in[1];
  const float* in_w     = (const float*)d_in[2];
  const float* in_b     = (const float*)d_in[3];
  const float* out_w    = (const float*)d_in[4];
  const float* out_b    = (const float*)d_in[5];
  const float* g1w      = (const float*)d_in[6];
  const float* g1b      = (const float*)d_in[7];
  const float* g2w      = (const float*)d_in[8];
  const float* g2b      = (const float*)d_in[9];
  const float* lnw      = (const float*)d_in[10];
  const float* lnb      = (const float*)d_in[11];

  const int D = 512;
  const size_t M = 128 * 256;            // 32768 rows
  char* ws = (char*)d_ws;
  size_t off = 0;
  auto carve = [&](size_t bytes) -> char* {
    char* p = ws + off; off += (bytes + 255) & ~(size_t)255; return p;
  };
  _Float16* Xh    = (_Float16*)carve(M * D * 2);
  _Float16* Wqkv  = (_Float16*)carve((size_t)3 * D * D * 2);
  _Float16* WoutH = (_Float16*)carve((size_t)D * D * 2);
  _Float16* G1h   = (_Float16*)carve((size_t)(D / 2) * D * 2);
  _Float16* Qh    = (_Float16*)carve(M * D * 2);          // (bh,256,64)
  _Float16* Kh    = (_Float16*)carve(M * D * 2);
  _Float16* Vt    = (_Float16*)carve(M * D * 2);          // (bh,64,256)
  _Float16* Ctxh  = (_Float16*)carve(M * D * 2);          // (m,512) f16
  float*    attF  = (float*)   carve(M * D * 4);          // attended fp32
  _Float16* attH  = (_Float16*)carve(M * D * 2);          // attended f16
  _Float16* Hgate = (_Float16*)carve(M * 256 * 2);        // relu hidden

  auto cvt = [&](const float* s, _Float16* d, size_t n) {
    f32_to_f16_kernel<<<(unsigned)((n + 255) / 256), 256, 0, stream>>>(s, d, (int)n);
  };
  cvt(features, Xh, M * D);
  cvt(in_w, Wqkv, (size_t)3 * D * D);
  cvt(out_w, WoutH, (size_t)D * D);
  cvt(g1w, G1h, (size_t)(D / 2) * D);

  gemm_wmma<<<dim3(1536 / BN, (unsigned)(M / BM)), 256, 0, stream>>>(
      Xh, Wqkv, in_b, (int)M, 1536, 512, 0, Qh, Kh, Vt);

  attn_flash<<<128 * 8 * 16, 32, 0, stream>>>(Qh, Kh, Vt, Ctxh);

  gemm_wmma<<<dim3(512 / BN, (unsigned)(M / BM)), 256, 0, stream>>>(
      Ctxh, WoutH, out_b, (int)M, 512, 512, 1, attF, attH, nullptr);

  gemm_wmma<<<dim3(256 / BN, (unsigned)(M / BM)), 256, 0, stream>>>(
      attH, G1h, g1b, (int)M, 256, 512, 2, Hgate, nullptr, nullptr);

  float* wOut = (float*)d_out + M * D;
  gate_topk<<<128, 256, 0, stream>>>(Hgate, g2w, g2b, sal, wOut);
  weight_ln<<<(unsigned)M, 128, 0, stream>>>(attF, wOut, lnw, lnb, (float*)d_out);
}
